// StandardMultiheadAttention_61220463837633
// MI455X (gfx1250) — matrix operand order repaired and checked by reading, not compiled, not run on hardware
//
#include <hip/hip_runtime.h>
#include <hip/hip_bf16.h>
#include <math.h>

// Problem constants (match reference)
#define BB 2
#define SS 2048
#define MM 1024
#define HH 16
#define KVH 4
#define DD 64
#define GG (HH / KVH)
#define KVDIM (KVH * DD)   // 256
#define ROWS (BB * SS)     // 4096

typedef __attribute__((ext_vector_type(16))) _Float16 v16h;
typedef __attribute__((ext_vector_type(8)))  float    v8f;

// ---------------------------------------------------------------------------
// Optional CDNA5 async global->LDS path (ASYNCcnt). Guarded: if the builtins
// are absent on this toolchain we fall back to synchronous LDS staging.
// Probe (round 3 diagnostic) showed the builtin takes pointers to 16-byte int
// vectors: (v4i* gaddr, v4i* lds, int offset, int cpol).
// ---------------------------------------------------------------------------
#if defined(__has_builtin)
#if __has_builtin(__builtin_amdgcn_global_load_async_to_lds_b128) && \
    __has_builtin(__builtin_amdgcn_s_wait_asynccnt)
#define HAVE_ASYNC_LDS 1
#endif
#endif
#ifndef HAVE_ASYNC_LDS
#define HAVE_ASYNC_LDS 0
#endif

#if HAVE_ASYNC_LDS
typedef __attribute__((__vector_size__(4 * sizeof(int)))) int async_v4i;
__device__ __forceinline__ void async_copy_b128(const _Float16* g, _Float16* l) {
    __builtin_amdgcn_global_load_async_to_lds_b128(
        (async_v4i*)g, (async_v4i*)l, /*offset=*/0, /*cpol=*/0);
}
template <int N>
__device__ __forceinline__ void wait_asynccnt() {
    __builtin_amdgcn_s_wait_asynccnt(N);
}
#endif

// ---------------------------------------------------------------------------
// WMMA fragment layouts (CDNA5 wave32, ISA 7.12.2)
//  A (16x32 f16): lane L (row M = L&15). lanes 0-15: K=0..15 in a[0..15],
//                 lanes 16-31: K=16..31 in a[0..15] (contiguous k order, as B).
//  B (32x16 f16): lane L: N = L&15; b[j]: K = j + (L<16 ? 0 : 16) (this code's order, not the ISA table order).
//  C/D (16x16 f32): lane L: N = L&15; c[r]: M = r + (L<16 ? 0 : 8).
// ---------------------------------------------------------------------------
__device__ __forceinline__ v8f wmma_f16(v16h a, v16h b, v8f c) {
    return __builtin_amdgcn_wmma_f32_16x16x32_f16(
        /*neg_a=*/false, a, /*neg_b=*/false, b,
        /*c_mod=*/(short)0, c, /*reuse_a=*/false, /*reuse_b=*/false);
}

// ---------------------------------------------------------------------------
// Generic GEMM: C[R x N] = A[R x K] * W[K x N] + bias[N]
// A: f32 or f16 row-major, W: f32 row-major, C: f32 row-major.
// Block tile 64x64, 4 waves (128 threads), K-step 64 (8 wmma / barrier).
// A/W staged through VALU (needs f32->f16 convert), with global prefetch
// of the next tiles. All dims are multiples of tile sizes (no bounds checks).
// ---------------------------------------------------------------------------
template <typename TA>
__global__ __launch_bounds__(128, 1)
void wmma_gemm_bias(const TA* __restrict__ A,
                    const float* __restrict__ W,
                    const float* __restrict__ bias,
                    float* __restrict__ C,
                    int R, int N, int K) {
    __shared__ _Float16 As[64][72];   // [row][k], padded rows
    __shared__ _Float16 Wt[64][72];   // [n][k], W tile transposed

    const int tid  = threadIdx.x;
    const int wave = tid >> 5;
    const int lane = tid & 31;
    const int brow = blockIdx.x * 64;
    const int bcol = blockIdx.y * 64;

    const int n0   = lane & 15;
    const int koff = (lane < 16) ? 0 : 16;
    const int hi   = (lane < 16) ? 0 : 8;

    v8f acc[4] = {v8f{}, v8f{}, v8f{}, v8f{}};

    for (int k0 = 0; k0 < K; k0 += 64) {
        // prefetch next tiles into cache (global_prefetch_b8)
        if (k0 + 64 < K) {
            const int pr = tid & 63;
            if (tid < 64)
                __builtin_prefetch(A + (size_t)(brow + pr) * K + (k0 + 64), 0, 1);
            else
                __builtin_prefetch(W + (size_t)(k0 + 64 + pr) * N + bcol, 0, 1);
        }
        // stage A tile 64x64 (32 elem / thread), convert to f16
        for (int i = tid; i < 64 * 64; i += 128) {
            const int r = i >> 6, c = i & 63;
            As[r][c] = (_Float16)(float)A[(size_t)(brow + r) * K + (k0 + c)];
        }
        // stage W tile 64x64 transposed (coalesced over n)
        for (int i = tid; i < 64 * 64; i += 128) {
            const int k = i >> 6, n = i & 63;
            Wt[n][k] = (_Float16)W[(size_t)(k0 + k) * N + (bcol + n)];
        }
        __syncthreads();

        // A fragments for this wave's 16 rows: k 0..31 and 32..63
        v16h a0, a1;
        {
            const int row = wave * 16 + (lane & 15);
#pragma unroll
            for (int j = 0; j < 8; ++j) {
                a0[j]     = As[row][koff + j];
                a0[8 + j] = As[row][koff + 8 + j];
                a1[j]     = As[row][32 + koff + j];
                a1[8 + j] = As[row][32 + koff + 8 + j];
            }
        }
#pragma unroll
        for (int g = 0; g < 4; ++g) {
            v16h b0, b1;
            const int n = g * 16 + n0;
#pragma unroll
            for (int j = 0; j < 16; ++j) {
                b0[j] = Wt[n][koff + j];
                b1[j] = Wt[n][32 + koff + j];
            }
            acc[g] = wmma_f16(a0, b0, acc[g]);
            acc[g] = wmma_f16(a1, b1, acc[g]);
        }
        __syncthreads();
    }

    // store with bias
    const int moff = (lane < 16) ? 0 : 8;
#pragma unroll
    for (int g = 0; g < 4; ++g) {
        const int col = bcol + g * 16 + n0;
        const float bv = bias[col];
#pragma unroll
        for (int r = 0; r < 8; ++r) {
            const int row = brow + wave * 16 + moff + r;
            C[(size_t)row * N + col] = acc[g][r] + bv;
        }
    }
}

// ---------------------------------------------------------------------------
// RoPE (half-split variant, matches reference) + f32 -> f16 convert.
// ---------------------------------------------------------------------------
__global__ void rope_cvt(const float* __restrict__ X, _Float16* __restrict__ Xh,
                         int nh, size_t total_pairs) {
    const size_t stride = (size_t)gridDim.x * blockDim.x;
    for (size_t idx = (size_t)blockIdx.x * blockDim.x + threadIdx.x;
         idx < total_pairs; idx += stride) {
        size_t t = idx;
        const int i = (int)(t & 31); t >>= 5;           // pair index in half=32
        const int head = (int)(t % nh); t /= nh;        // head
        const size_t rr = t;                            // row in [0, B*S)
        const int s = (int)(rr & (SS - 1));             // position
        const float inv_freq = __powf(10000.0f, -(float)i * (1.0f / 32.0f));
        const float ang = (float)s * inv_freq;
        float c, sn;
        __sincosf(ang, &sn, &c);
        const size_t base = rr * (size_t)(nh * DD) + (size_t)head * DD;
        const float x1 = X[base + i];
        const float x2 = X[base + 32 + i];
        Xh[base + i]      = (_Float16)(x1 * c - x2 * sn);
        Xh[base + 32 + i] = (_Float16)(x1 * sn + x2 * c);
    }
}

__global__ void cvt_f32_f16(const float* __restrict__ X, _Float16* __restrict__ Xh,
                            size_t n) {
    const size_t stride = (size_t)gridDim.x * blockDim.x;
    for (size_t i = (size_t)blockIdx.x * blockDim.x + threadIdx.x; i < n; i += stride)
        Xh[i] = (_Float16)X[i];
}

// ---------------------------------------------------------------------------
// Flash attention forward (non-causal, GQA).
// grid = (S/64, H, B), block = 128 (4 waves). Each wave owns 16 q rows.
// KV processed 64 rows per step: QK^T (8 wmma), online softmax (max via
// 16-lane shfl; row-SUM via WMMA against an all-ones B fragment), P@V (8
// wmma) + 2 row-sum wmma => 18 wmma per barrier pair.
// K tiles are double-buffered and DMA'd via GLOBAL_LOAD_ASYNC_TO_LDS_B128
// (ASYNCcnt pipeline) when the toolchain exposes the builtin; V tiles are
// transposed through the VALU (async DMA cannot transpose).
// ---------------------------------------------------------------------------
__global__ __launch_bounds__(128, 1)
void attn_fwd(const _Float16* __restrict__ qh,
              const _Float16* __restrict__ kh,
              const _Float16* __restrict__ vh,
              _Float16* __restrict__ ctx) {
    __shared__ _Float16 Ks[2][64][72];    // [buf][kv][d], 16B-aligned rows (144B pitch)
    __shared__ _Float16 Vt[64][72];       // [d][kv]
    __shared__ _Float16 Ps[4][16][72];    // per-wave P tile [q][kv 0..63]

    const int tid  = threadIdx.x;
    const int wave = tid >> 5;
    const int lane = tid & 31;
    const int h    = blockIdx.y;
    const int b    = blockIdx.z;
    const int kvh  = h >> 2;                       // G = 4
    const int qbase = blockIdx.x * 64 + wave * 16;

    const _Float16* qptr = qh + ((size_t)b * SS) * MM    + (size_t)h   * DD;
    const _Float16* kptr = kh + ((size_t)b * SS) * KVDIM + (size_t)kvh * DD;
    const _Float16* vptr = vh + ((size_t)b * SS) * KVDIM + (size_t)kvh * DD;

    const int n0   = lane & 15;
    const int koff = (lane < 16) ? 0 : 16;
    const int hi   = (lane < 16) ? 0 : 8;
    const int moff = (lane < 16) ? 0 : 8;

    // Q fragments (d 0..31 / 32..63) loaded once, straight into A layout.
    v16h qa0, qa1;
    {
        const int row = qbase + (lane & 15);
        const _Float16* qr = qptr + (size_t)row * MM;
#pragma unroll
        for (int j = 0; j < 8; ++j) {
            qa0[j]     = qr[koff + j];
            qa0[8 + j] = qr[koff + 8 + j];
            qa1[j]     = qr[32 + koff + j];
            qa1[8 + j] = qr[32 + koff + 8 + j];
        }
    }

    // all-ones B fragment: row-sum of P via WMMA
    v16h ones;
#pragma unroll
    for (int j = 0; j < 16; ++j) ones[j] = (_Float16)1.0f;

    float mrow[8];
    v8f o[4] = {v8f{}, v8f{}, v8f{}, v8f{}};
    v8f sacc = v8f{};   // running row sums (broadcast across the 16 N lanes)
#pragma unroll
    for (int r = 0; r < 8; ++r) mrow[r] = -1e30f;

    const float scale = 0.125f;  // 1/sqrt(64)

#if HAVE_ASYNC_LDS
    // kick off the async DMA of K tile 0 (4 x b128 chunks per thread)
#pragma unroll
    for (int q = 0; q < 4; ++q) {
        const int c  = tid + 128 * q;         // 512 chunks = 64 rows x 8
        const int kv = c >> 3, part = c & 7;
        async_copy_b128(kptr + (size_t)kv * KVDIM + part * 8, &Ks[0][kv][part * 8]);
    }
#endif

    int cur = 0;
    for (int t = 0; t < SS; t += 64) {
        const bool more = (t + 64) < SS;

        // prefetch next V rows (global_prefetch_b8)
        if (more && tid < 64)
            __builtin_prefetch(vptr + (size_t)(t + 64 + tid) * KVDIM, 0, 1);

        // stage V transposed (VALU path): 64 kv x 64 d
        for (int i = tid; i < 64 * 64; i += 128) {
            const int kv = i >> 6, d = i & 63;
            Vt[d][kv] = vptr[(size_t)(t + kv) * KVDIM + d];
        }

#if HAVE_ASYNC_LDS
        // issue async DMA of the NEXT K tile into the other buffer, then
        // retire the current tile's 4 chunks (async loads retire in order).
        if (more) {
#pragma unroll
            for (int q = 0; q < 4; ++q) {
                const int c  = tid + 128 * q;
                const int kv = c >> 3, part = c & 7;
                async_copy_b128(kptr + (size_t)(t + 64 + kv) * KVDIM + part * 8,
                                &Ks[cur ^ 1][kv][part * 8]);
            }
            wait_asynccnt<4>();
        } else {
            wait_asynccnt<0>();
        }
#else
        // synchronous staging fallback
        for (int i = tid; i < 64 * 64; i += 128) {
            const int kv = i >> 6, d = i & 63;
            Ks[cur][kv][d] = kptr[(size_t)(t + kv) * KVDIM + d];
        }
#endif
        __syncthreads();

        // ----- scores: four 16x16 tiles over kv, K-dim 64 -> 8 wmma -----
        v8f s[4];
#pragma unroll
        for (int tt = 0; tt < 4; ++tt) {
            v16h b0, b1;
            const int kvr = tt * 16 + n0;
#pragma unroll
            for (int j = 0; j < 16; ++j) {
                b0[j] = Ks[cur][kvr][koff + j];           // d 0..31
                b1[j] = Ks[cur][kvr][32 + koff + j];      // d 32..63
            }
            s[tt] = v8f{};
            s[tt] = wmma_f16(qa0, b0, s[tt]);
            s[tt] = wmma_f16(qa1, b1, s[tt]);
        }

        // ----- online softmax: max via 16-lane shfl, sum via WMMA later -----
#pragma unroll
        for (int r = 0; r < 8; ++r) {
            const float v0 = s[0][r] * scale;
            const float v1 = s[1][r] * scale;
            const float v2 = s[2][r] * scale;
            const float v3 = s[3][r] * scale;
            float rm = fmaxf(fmaxf(v0, v1), fmaxf(v2, v3));
#pragma unroll
            for (int off = 1; off < 16; off <<= 1)
                rm = fmaxf(rm, __shfl_xor(rm, off, 32));
            const float mn   = fmaxf(mrow[r], rm);
            const float corr = __expf(mrow[r] - mn);
            mrow[r] = mn;
            // rescale running output and running sum (same per-row factor)
            sacc[r] *= corr;
#pragma unroll
            for (int g = 0; g < 4; ++g) o[g][r] *= corr;
            // exponentials -> per-wave LDS tile in [q][kv] order
            Ps[wave][r + moff][n0]      = (_Float16)__expf(v0 - mn);
            Ps[wave][r + moff][16 + n0] = (_Float16)__expf(v1 - mn);
            Ps[wave][r + moff][32 + n0] = (_Float16)__expf(v2 - mn);
            Ps[wave][r + moff][48 + n0] = (_Float16)__expf(v3 - mn);
        }

        // reload P as two A fragments (kv 0..31, 32..63); same-wave LDS dep,
        // DS ops are in-order per wave and the compiler inserts s_wait_dscnt.
        v16h pa0, pa1;
        {
            const int row = lane & 15;
#pragma unroll
            for (int j = 0; j < 8; ++j) {
                pa0[j]     = Ps[wave][row][koff + j];
                pa0[8 + j] = Ps[wave][row][koff + 8 + j];
                pa1[j]     = Ps[wave][row][32 + koff + j];
                pa1[8 + j] = Ps[wave][row][32 + koff + 8 + j];
            }
        }

        // running row sums: sacc += P @ ones  (2 wmma)
        sacc = wmma_f16(pa0, ones, sacc);
        sacc = wmma_f16(pa1, ones, sacc);

        // ----- O += P (16x64) @ V (64x64) : 8 wmma -----
#pragma unroll
        for (int g = 0; g < 4; ++g) {
            v16h vb0, vb1;
            const int d = g * 16 + n0;
#pragma unroll
            for (int j = 0; j < 16; ++j) {
                vb0[j] = Vt[d][koff + j];            // kv 0..31
                vb1[j] = Vt[d][32 + koff + j];       // kv 32..63
            }
            o[g] = wmma_f16(pa0, vb0, o[g]);
            o[g] = wmma_f16(pa1, vb1, o[g]);
        }
        __syncthreads();   // all waves done with Ks[cur]/Vt before restaging
        cur ^= 1;
    }

    // normalize and store ctx (f16)
#pragma unroll
    for (int r = 0; r < 8; ++r) {
        const float inv = 1.0f / sacc[r];
        const int qrow = qbase + r + moff;
        const size_t base = ((size_t)b * SS + qrow) * MM + (size_t)h * DD;
#pragma unroll
        for (int g = 0; g < 4; ++g)
            ctx[base + g * 16 + n0] = (_Float16)(o[g][r] * inv);
    }
}

// ---------------------------------------------------------------------------
// Host-side orchestration
// ---------------------------------------------------------------------------
extern "C" void kernel_launch(void* const* d_in, const int* in_sizes, int n_in,
                              void* d_out, int out_size, void* d_ws, size_t ws_size,
                              hipStream_t stream) {
    const float* seqs   = (const float*)d_in[0];
    const float* keys   = (const float*)d_in[1];
    const float* values = (const float*)d_in[2];
    const float* Wq = (const float*)d_in[3];
    const float* bq = (const float*)d_in[4];
    const float* Wk = (const float*)d_in[5];
    const float* bk = (const float*)d_in[6];
    const float* Wv = (const float*)d_in[7];
    const float* bv = (const float*)d_in[8];
    const float* Wo = (const float*)d_in[9];
    const float* bo = (const float*)d_in[10];
    float* out = (float*)d_out;

    // workspace layout (all within d_ws), ~44 MB total
    char* ws = (char*)d_ws;
    float*     Qf = (float*)ws;                        ws += (size_t)ROWS * MM    * 4;
    float*     Kf = (float*)ws;                        ws += (size_t)ROWS * KVDIM * 4;
    float*     Vf = (float*)ws;                        ws += (size_t)ROWS * KVDIM * 4;
    _Float16*  qh = (_Float16*)ws;                     ws += (size_t)ROWS * MM    * 2;
    _Float16*  kh = (_Float16*)ws;                     ws += (size_t)ROWS * KVDIM * 2;
    _Float16*  vh = (_Float16*)ws;                     ws += (size_t)ROWS * KVDIM * 2;
    _Float16*  ch = (_Float16*)ws;                     ws += (size_t)ROWS * MM    * 2;
    (void)ws_size; (void)in_sizes; (void)n_in; (void)out_size;

    // 1) projections (f32 in, f16 WMMA, f32 out + bias)
    wmma_gemm_bias<float><<<dim3(ROWS / 64, MM / 64),    128, 0, stream>>>(seqs,   Wq, bq, Qf, ROWS, MM,    MM);
    wmma_gemm_bias<float><<<dim3(ROWS / 64, KVDIM / 64), 128, 0, stream>>>(keys,   Wk, bk, Kf, ROWS, KVDIM, MM);
    wmma_gemm_bias<float><<<dim3(ROWS / 64, KVDIM / 64), 128, 0, stream>>>(values, Wv, bv, Vf, ROWS, KVDIM, MM);

    // 2) RoPE(q,k) + convert to f16; v converted directly
    {
        const size_t qpairs = (size_t)ROWS * HH * 32;
        const size_t kpairs = (size_t)ROWS * KVH * 32;
        rope_cvt<<<(int)((qpairs + 255) / 256), 256, 0, stream>>>(Qf, qh, HH, qpairs);
        rope_cvt<<<(int)((kpairs + 255) / 256), 256, 0, stream>>>(Kf, kh, KVH, kpairs);
        const size_t vn = (size_t)ROWS * KVDIM;
        cvt_f32_f16<<<(int)((vn + 255) / 256), 256, 0, stream>>>(Vf, vh, vn);
    }

    // 3) flash attention -> ctx (f16)
    attn_fwd<<<dim3(SS / 64, HH, BB), 128, 0, stream>>>(qh, kh, vh, ch);

    // 4) output projection: out = ctx @ Wo + bo (f32 out)
    wmma_gemm_bias<_Float16><<<dim3(ROWS / 64, MM / 64), 128, 0, stream>>>(ch, Wo, bo, out, ROWS, MM, MM);
}
